// BitSAE_34746285425464
// MI455X (gfx1250) — 8-bit matrix operand order repaired and checked by reading, compiled, not run on hardware
//
#include <hip/hip_runtime.h>
#include <stdint.h>

typedef __attribute__((ext_vector_type(8))) int v8i;

#define EPSQ 1e-5f

__global__ void BitSAE_zero_sums(float* p) {
  if (threadIdx.x < 2) p[threadIdx.x] = 0.0f;
}

__global__ void BitSAE_abs_sum(const float* __restrict__ w, size_t n,
                               float* __restrict__ acc) {
  __shared__ float red[256];
  const size_t stride = (size_t)gridDim.x * blockDim.x;
  float s = 0.0f;
  for (size_t i = (size_t)blockIdx.x * blockDim.x + threadIdx.x; i < n; i += stride)
    s += fabsf(w[i]);
  red[threadIdx.x] = s;
  __syncthreads();
  for (int off = 128; off > 0; off >>= 1) {
    if ((int)threadIdx.x < off) red[threadIdx.x] += red[threadIdx.x + off];
    __syncthreads();
  }
  if (threadIdx.x == 0) atomicAdd(acc, red[0]);
}

__global__ void BitSAE_quant_weight(const float* __restrict__ w, size_t n,
                                    const float* __restrict__ sum, float inv_count,
                                    int8_t* __restrict__ q) {
  const float mean  = fmaxf(sum[0] * inv_count, EPSQ);
  const float scale = 1.0f / mean;
  const size_t stride = (size_t)gridDim.x * blockDim.x;
  for (size_t i = (size_t)blockIdx.x * blockDim.x + threadIdx.x; i < n; i += stride) {
    float v = rintf(w[i] * scale);
    v = fminf(fmaxf(v, -1.0f), 1.0f);
    q[i] = (int8_t)v;
  }
}

__global__ void BitSAE_quant_rows(const float* __restrict__ X,
                                  const float* __restrict__ sub,
                                  int8_t* __restrict__ Q,
                                  float* __restrict__ scl_inv,
                                  int ncols) {
  __shared__ float red[256];
  const int row = blockIdx.x;
  const float* xr = X + (size_t)row * ncols;
  float mx = 0.0f;
  for (int c = threadIdx.x; c < ncols; c += blockDim.x) {
    float v = xr[c] - (sub ? sub[c] : 0.0f);
    mx = fmaxf(mx, fabsf(v));
  }
  red[threadIdx.x] = mx;
  __syncthreads();
  for (int off = 128; off > 0; off >>= 1) {
    if ((int)threadIdx.x < off)
      red[threadIdx.x] = fmaxf(red[threadIdx.x], red[threadIdx.x + off]);
    __syncthreads();
  }
  const float m = fmaxf(red[0], EPSQ);
  const float scale = 127.0f / m;
  if (threadIdx.x == 0) scl_inv[row] = m / 127.0f;
  int8_t* qr = Q + (size_t)row * ncols;
  for (int c = threadIdx.x; c < ncols; c += blockDim.x) {
    float v = xr[c] - (sub ? sub[c] : 0.0f);
    float q = fminf(fmaxf(rintf(v * scale), -128.0f), 127.0f);
    qr[c] = (int8_t)q;
  }
}

template <bool RELU>
__global__ __launch_bounds__(32)
void BitSAE_gemm_iu8(const int8_t* __restrict__ Aq,
                     const int8_t* __restrict__ Wq,
                     const float*  __restrict__ a_scl,
                     const float*  __restrict__ wsum,
                     float inv_count,
                     const float*  __restrict__ bias,
                     float*        __restrict__ C,
                     int M, int N, int K) {
  const int lane = threadIdx.x & 31;
  const int half = lane >> 4;
  const int lm   = lane & 15;
  const int m0 = blockIdx.y * 64;
  const int n0 = blockIdx.x * 64;
  const float wscale = fmaxf(wsum[0] * inv_count, EPSQ);

  v8i acc[4][4];
#pragma unroll
  for (int i = 0; i < 4; ++i)
#pragma unroll
    for (int j = 0; j < 4; ++j) {
      v8i z = {0, 0, 0, 0, 0, 0, 0, 0};
      acc[i][j] = z;
    }

  const int8_t* arow[4];
  const int8_t* brow[4];
#pragma unroll
  for (int i = 0; i < 4; ++i)
    arow[i] = Aq + (size_t)(m0 + i * 16 + lm) * K + half * 16;
#pragma unroll
  for (int j = 0; j < 4; ++j)
    brow[j] = Wq + (size_t)(n0 + j * 16 + lm) * K + half * 16;

  for (int k = 0; k < K; k += 64) {
    v8i a[4];
#pragma unroll
    for (int i = 0; i < 4; ++i) {
      const int8_t* p = arow[i] + k;
      int2 t0 = *(const int2*)(p);
      int2 t1 = *(const int2*)(p + 8);
      int2 t2 = *(const int2*)(p + 32);
      int2 t3 = *(const int2*)(p + 40);
      v8i av = {t0.x, t0.y, t1.x, t1.y, t2.x, t2.y, t3.x, t3.y};
      a[i] = av;
    }
#pragma unroll
    for (int j = 0; j < 4; ++j) {
      const int8_t* p = brow[j] + k;
      __builtin_prefetch(p + 64, 0, 0);
      int4 u0 = *(const int4*)(p);
      int4 u1 = *(const int4*)(p + 32);
      v8i bv = {u0.x, u0.y, u0.z, u0.w, u1.x, u1.y, u1.z, u1.w};
#pragma unroll
      for (int i = 0; i < 4; ++i)
        acc[i][j] = __builtin_amdgcn_wmma_i32_16x16x64_iu8(
             true, a[i],  true, bv, acc[i][j],
             false,  false);
    }
  }

#pragma unroll
  for (int j = 0; j < 4; ++j) {
    const int n = n0 + j * 16 + lm;
    const float bn = bias[n];
#pragma unroll
    for (int i = 0; i < 4; ++i) {
#pragma unroll
      for (int v = 0; v < 8; ++v) {
        const int m = m0 + i * 16 + half * 8 + v;
        float val = (float)acc[i][j][v] * a_scl[m] * wscale + bn;
        if (RELU) val = fmaxf(val, 0.0f);
        C[(size_t)m * N + n] = val;
      }
    }
  }
}

extern "C" void kernel_launch(void* const* d_in, const int* in_sizes, int n_in,
                              void* d_out, int out_size, void* d_ws, size_t ws_size,
                              hipStream_t stream) {
  (void)n_in; (void)out_size; (void)ws_size;
  const float* x     = (const float*)d_in[0];
  const float* enc_w = (const float*)d_in[1];
  const float* enc_b = (const float*)d_in[2];
  const float* dec_w = (const float*)d_in[3];
  const float* dec_b = (const float*)d_in[4];

  const int F = in_sizes[4];
  const int H = in_sizes[2];
  const int B = in_sizes[0] / F;
  const size_t WN = (size_t)H * F;
  const float inv_count = 1.0f / (float)WN;

  char* ws = (char*)d_ws;
  size_t o = 0;
  auto take = [&](size_t bytes) -> void* {
    void* p = ws + o;
    o += (bytes + 255) & ~(size_t)255;
    return p;
  };
  float*  sums  = (float*) take(2 * sizeof(float));
  float*  x_scl = (float*) take((size_t)B * 4);
  float*  f_scl = (float*) take((size_t)B * 4);
  int8_t* xq    = (int8_t*)take((size_t)B * F);
  int8_t* fq    = (int8_t*)take((size_t)B * H);
  int8_t* ewq   = (int8_t*)take(WN);
  int8_t* dwq   = (int8_t*)take(WN);

  float* xhat = (float*)d_out;
  float* fout = (float*)d_out + (size_t)B * F;

  BitSAE_zero_sums<<<1, 32, 0, stream>>>(sums);
  BitSAE_abs_sum<<<2048, 256, 0, stream>>>(enc_w, WN, sums + 0);
  BitSAE_abs_sum<<<2048, 256, 0, stream>>>(dec_w, WN, sums + 1);
  BitSAE_quant_weight<<<4096, 256, 0, stream>>>(enc_w, WN, sums + 0, inv_count, ewq);
  BitSAE_quant_weight<<<4096, 256, 0, stream>>>(dec_w, WN, sums + 1, inv_count, dwq);
  BitSAE_quant_rows<<<B, 256, 0, stream>>>(x, dec_b, xq, x_scl, F);
  dim3 g1(H / 64, B / 64);
  BitSAE_gemm_iu8<true><<<g1, 32, 0, stream>>>(xq, ewq, x_scl, sums + 0, inv_count,
                                               enc_b, fout, B, H, F);
  BitSAE_quant_rows<<<B, 256, 0, stream>>>(fout, nullptr, fq, f_scl, H);
  dim3 g2(F / 64, B / 64);
  BitSAE_gemm_iu8<false><<<g2, 32, 0, stream>>>(fq, dwq, f_scl, sums + 1, inv_count,
                                                dec_b, xhat, B, F, H);
}
